// RWKV_14774687498336
// MI455X (gfx1250) — matrix operand order repaired and checked by reading, not compiled, not run on hardware
//
#include <hip/hip_runtime.h>
#include <hip/hip_bf16.h>

// ---------------------------------------------------------------------------
// RWKV-v5 forward for MI455X (gfx1250, wave32, WMMA).
// All dense layers run through v_wmma_f32_16x16x32_bf16 with:
//   - token-shift "mix" fused into the GEMM A-tile staging,
//   - activation (relu^2 / silu / sigmoid), gate-multiply and residual-add
//     fused into the GEMM epilogue.
// wkv recurrence: one workgroup per (b,h), 64x64 state held in VGPRs.
// ---------------------------------------------------------------------------

typedef __attribute__((ext_vector_type(16))) __bf16 v16bf;
typedef __attribute__((ext_vector_type(8)))  float  v8f;

#define B_SZ 2
#define T_SZ 1024
#define C_SZ 2048
#define H_SZ 32
#define K_HD 64
#define F_SZ 7168
#define V_SZ 32000
#define L_SZ 2
#define M_SZ (B_SZ * T_SZ)

#define TILE_M 128
#define TILE_N 128
#define TILE_K 32
#define LDST   40   // padded bf16 elems per LDS row (80B -> 16B-aligned, bank-clean)

enum { ACT_NONE = 0, ACT_RELU2 = 1, ACT_SILU = 2, ACT_SIG = 3 };

static __device__ __forceinline__ unsigned short f2bf(float f) {
  unsigned u = __float_as_uint(f);
  u += 0x7FFFu + ((u >> 16) & 1u);          // round-to-nearest-even
  return (unsigned short)(u >> 16);
}
static __device__ __forceinline__ float bf2f(unsigned short h) {
  return __uint_as_float(((unsigned)h) << 16);
}

union FragU { v16bf v; uint4 q[2]; };

// A fragment (16x32 MxK, bf16): lane<16 -> row=lane, K in {0..7, 16..23};
// lane>=16 -> row=lane-16, K in {8..15, 24..31}. Two b128 LDS loads.
static __device__ __forceinline__ v16bf load_frag_A(const unsigned short* lds,
                                                    int r0, int lane) {
  int hi = lane >> 4;
  const unsigned short* p = lds + (size_t)(r0 + (lane & 15)) * LDST;
  FragU f;
  f.q[0] = *(const uint4*)(p + hi * 8);
  f.q[1] = *(const uint4*)(p + 16 + hi * 8);
  return f.v;
}
// B fragment (32x16 KxN, bf16): lane holds column n=lane&15, K in {8*hi..8*hi+7, 16+8*hi..16+8*hi+7}.
static __device__ __forceinline__ v16bf load_frag_B(const unsigned short* lds,
                                                    int n0, int lane) {
  int hi = lane >> 4;
  const unsigned short* p = lds + (size_t)(n0 + (lane & 15)) * LDST;
  FragU f;
  f.q[0] = *(const uint4*)(p + hi * 8);
  f.q[1] = *(const uint4*)(p + 16 + hi * 8);
  return f.v;
}

// ---------------------------------------------------------------------------
// C[m][n] = act( sum_k A'[m][k] * W[n][k] ) * (mulC ? mulC[m][n] : 1)
//           + (addC ? addC[m][n] : 0)
// A'[m][k] = mixC ? A[m][k]*mixC[k] + prev[m][k]*(1-mixC[k]) : A[m][k]
// prev = shift-state row for t==0, else previous row of A (token shift).
// M,N multiples of 128; K multiple of 32.
// ---------------------------------------------------------------------------
__global__ __launch_bounds__(256) void gemm_nt(
    const float* __restrict__ A, const float* __restrict__ W,
    float* __restrict__ C,
    const float* __restrict__ mixC, const float* __restrict__ shiftS, int Tseq,
    const float* __restrict__ mulC, const float* __restrict__ addC,
    int M, int N, int K, int act) {
  __shared__ alignas(16) unsigned short As[TILE_M * LDST];
  __shared__ alignas(16) unsigned short Bs[TILE_N * LDST];

  const int tid   = threadIdx.x;
  const int lane  = tid & 31;
  const int wave  = tid >> 5;
  const int waveM = wave >> 1;   // 0..3 -> 32-row band
  const int waveN = wave & 1;    // 0..1 -> 64-col band
  const int rowBase = blockIdx.y * TILE_M;
  const int colBase = blockIdx.x * TILE_N;

  v8f acc[2][4];
  const v8f vzero = {0.f, 0.f, 0.f, 0.f, 0.f, 0.f, 0.f, 0.f};
#pragma unroll
  for (int mt = 0; mt < 2; ++mt)
#pragma unroll
    for (int nt = 0; nt < 4; ++nt) acc[mt][nt] = vzero;

  const int ldRow = tid >> 3;        // 0..31
  const int ldCol = (tid & 7) * 4;   // float4 column

  for (int k0 = 0; k0 < K; k0 += TILE_K) {
    // ---- stage A (with fused token-shift mix) and B tiles, fp32 -> bf16 ----
#pragma unroll
    for (int p = 0; p < 4; ++p) {
      int r = p * 32 + ldRow;
      {
        int gr = rowBase + r;
        int gk = k0 + ldCol;
        const float* ap = A + (size_t)gr * K + gk;
        float4 a = *(const float4*)ap;
        if (mixC) {
          float4 m = *(const float4*)(mixC + gk);
          int t = gr % Tseq;
          const float* pp = (t == 0) ? (shiftS + (size_t)(gr / Tseq) * K + gk)
                                     : (ap - K);
          float4 pv = *(const float4*)pp;
          a.x = a.x * m.x + pv.x * (1.f - m.x);
          a.y = a.y * m.y + pv.y * (1.f - m.y);
          a.z = a.z * m.z + pv.z * (1.f - m.z);
          a.w = a.w * m.w + pv.w * (1.f - m.w);
        }
        if (k0 + TILE_K < K) __builtin_prefetch(ap + TILE_K, 0, 1);
        unsigned short* d = As + (size_t)r * LDST + ldCol;
        d[0] = f2bf(a.x); d[1] = f2bf(a.y); d[2] = f2bf(a.z); d[3] = f2bf(a.w);
      }
      {
        int gr = colBase + r;
        const float* wp = W + (size_t)gr * K + k0 + ldCol;
        float4 b = *(const float4*)wp;
        if (k0 + TILE_K < K) __builtin_prefetch(wp + TILE_K, 0, 1);
        unsigned short* d = Bs + (size_t)r * LDST + ldCol;
        d[0] = f2bf(b.x); d[1] = f2bf(b.y); d[2] = f2bf(b.z); d[3] = f2bf(b.w);
      }
    }
    __syncthreads();

    v16bf af[2], bf[4];
#pragma unroll
    for (int mt = 0; mt < 2; ++mt)
      af[mt] = load_frag_A(As, waveM * 32 + mt * 16, lane);
#pragma unroll
    for (int nt = 0; nt < 4; ++nt)
      bf[nt] = load_frag_B(Bs, waveN * 64 + nt * 16, lane);
#pragma unroll
    for (int mt = 0; mt < 2; ++mt)
#pragma unroll
      for (int nt = 0; nt < 4; ++nt)
        acc[mt][nt] = __builtin_amdgcn_wmma_f32_16x16x32_bf16(
            false, af[mt], false, bf[nt], (short)0, acc[mt][nt], false, false);
    __syncthreads();
  }

  // ---- epilogue: act, gate-mul, residual-add, store ----
  const int hi   = lane >> 4;
  const int ln16 = lane & 15;
#pragma unroll
  for (int mt = 0; mt < 2; ++mt) {
#pragma unroll
    for (int nt = 0; nt < 4; ++nt) {
      int gn     = colBase + waveN * 64 + nt * 16 + ln16;
      int gmBase = rowBase + waveM * 32 + mt * 16 + 8 * hi;
#pragma unroll
      for (int r = 0; r < 8; ++r) {
        int gm  = gmBase + r;
        float x = acc[mt][nt][r];
        if (act == ACT_RELU2)      { float t = fmaxf(x, 0.f); x = t * t; }
        else if (act == ACT_SILU)  { x = x / (1.f + __expf(-x)); }
        else if (act == ACT_SIG)   { x = 1.f / (1.f + __expf(-x)); }
        size_t o = (size_t)gm * N + gn;
        if (mulC) x *= mulC[o];
        if (addC) x += addC[o];
        C[o] = x;
      }
    }
  }
}

// ---------------------------------------------------------------------------
__global__ void emb_kernel(const int* __restrict__ idx,
                           const float* __restrict__ emb,
                           float* __restrict__ out, int C) {
  size_t i = (size_t)blockIdx.x * 256 + threadIdx.x;
  int row = (int)(i / C), c = (int)(i % C);
  out[i] = emb[(size_t)idx[row] * C + c];
}

// LayerNorm over C=2048, one block (256 threads) per row; values cached in
// registers so in==out is safe.
__global__ __launch_bounds__(256) void ln_kernel(
    const float* __restrict__ in, const float* __restrict__ w,
    const float* __restrict__ b, float* __restrict__ out, int C, float eps) {
  const int row = blockIdx.x;
  const int tid = threadIdx.x;
  const float* p = in + (size_t)row * C;
  float vals[8];
  float s = 0.f;
#pragma unroll
  for (int i = 0; i < 8; ++i) { float v = p[tid + i * 256]; vals[i] = v; s += v; }
  __shared__ float red[256];
  red[tid] = s; __syncthreads();
  for (int off = 128; off; off >>= 1) {
    if (tid < off) red[tid] += red[tid + off];
    __syncthreads();
  }
  float mean = red[0] / C; __syncthreads();
  float s2 = 0.f;
#pragma unroll
  for (int i = 0; i < 8; ++i) { float d = vals[i] - mean; s2 += d * d; }
  red[tid] = s2; __syncthreads();
  for (int off = 128; off; off >>= 1) {
    if (tid < off) red[tid] += red[tid + off];
    __syncthreads();
  }
  float inv = rsqrtf(red[0] / C + eps);
  float* q = out + (size_t)row * C;
#pragma unroll
  for (int i = 0; i < 8; ++i) {
    int c = tid + i * 256;
    q[c] = (vals[i] - mean) * inv * w[c] + b[c];
  }
}

// wkv recurrence: block = (b,h); thread j owns output dim j and state row
// S[j][0..63] (kept in VGPRs via full unroll). r,k,w,u broadcast through LDS.
__global__ __launch_bounds__(64) void wkv_kernel(
    const float* __restrict__ r, const float* __restrict__ k,
    const float* __restrict__ v, const float* __restrict__ w,
    const float* __restrict__ u, const float* __restrict__ s0,
    float* __restrict__ out, int Tn, int Hn, int Cn) {
  const int b = blockIdx.x / Hn, h = blockIdx.x % Hn;
  const int j = threadIdx.x;
  float S[K_HD];
  const float* sp = s0 + (((size_t)b * Hn + h) * K_HD + j) * K_HD;  // S[j][i]
#pragma unroll
  for (int i = 0; i < K_HD; ++i) S[i] = sp[i];
  __shared__ float rs[K_HD], ks[K_HD], ws[K_HD], us[K_HD];
  ws[j] = w[h * K_HD + j];
  us[j] = u[h * K_HD + j];
  __syncthreads();
  for (int t = 0; t < Tn; ++t) {
    size_t base = ((size_t)b * Tn + t) * Cn + h * K_HD;
    rs[j] = r[base + j];
    ks[j] = k[base + j];
    __syncthreads();
    float vj = v[base + j];
    float acc = 0.f, d = 0.f;
#pragma unroll
    for (int i = 0; i < K_HD; ++i) {
      float ri = rs[i], ki = ks[i];
      acc += ri * S[i];                 // r_t . S[j][:]
      d   += us[i] * ri * ki;           // u-diag bonus (same for all j)
      S[i] = ws[i] * S[i] + ki * vj;    // state update
    }
    out[base + j] = acc + d * vj;
    __syncthreads();
  }
}

// per-head GroupNorm (after bf16 round-trip) fused with SiLU-gate multiply.
__global__ __launch_bounds__(64) void gn_gate(
    const float* __restrict__ att, const float* __restrict__ lnw,
    const float* __restrict__ lnb, const float* __restrict__ gate,
    float* __restrict__ out, float eps, int H, int C) {
  const int row = blockIdx.x / H, h = blockIdx.x % H;
  const int j = threadIdx.x;
  size_t base = (size_t)row * C + h * K_HD;
  float x = bf2f(f2bf(att[base + j]));     // bf16 round trip as in reference
  __shared__ float red[K_HD];
  red[j] = x; __syncthreads();
  for (int off = 32; off; off >>= 1) {
    if (j < off) red[j] += red[j + off];
    __syncthreads();
  }
  float mean = red[0] * (1.f / 64.f); __syncthreads();
  float d = x - mean;
  red[j] = d * d; __syncthreads();
  for (int off = 32; off; off >>= 1) {
    if (j < off) red[j] += red[j + off];
    __syncthreads();
  }
  float inv = rsqrtf(red[0] * (1.f / 64.f) + eps);
  out[base + j] = (d * inv * lnw[h * K_HD + j] + lnb[h * K_HD + j]) * gate[base + j];
}

// ---------------------------------------------------------------------------
// Input leaf order: jax sorted-key tree flatten of setup_inputs():
//   0 idx, 1 last_shift_states, 2 last_wkv_states, then params:
//   blocks[l] (22 leaves each, keys sorted):
//     att: +0 gate,+1 key,+2 ln_x_b,+3 ln_x_w,+4 output,+5 receptance,
//          +6 time_decay,+7 time_faaaa,+8 tm_g,+9 tm_k,+10 tm_r,+11 tm_v,+12 value
//     ffn: +13 key,+14 receptance,+15 tm_k,+16 tm_r,+17 value
//     +18 ln1_b,+19 ln1_w,+20 ln2_b,+21 ln2_w
//   then emb, head, ln_in_b, ln_in_w, ln_out_b, ln_out_w.
// ---------------------------------------------------------------------------
extern "C" void kernel_launch(void* const* d_in, const int* in_sizes, int n_in,
                              void* d_out, int out_size, void* d_ws,
                              size_t ws_size, hipStream_t stream) {
  (void)in_sizes; (void)n_in; (void)out_size; (void)ws_size;

  auto in = [&](int i) { return (const float*)d_in[i]; };
  const int* idx  = (const int*)d_in[0];
  const float* lss  = in(1);
  const float* wkvS = in(2);
  const int I_EMB = 3 + 22 * L_SZ;           // 47
  const int I_HEAD = I_EMB + 1;
  const int I_LNIN_B = I_EMB + 2, I_LNIN_W = I_EMB + 3;
  const int I_LNOUT_B = I_EMB + 4, I_LNOUT_W = I_EMB + 5;

  // workspace layout (fp32)
  const size_t MC = (size_t)M_SZ * C_SZ;
  float* X   = (float*)d_ws;
  float* XLN = X + MC;
  float* Rb  = XLN + MC;
  float* Kb  = Rb + MC;
  float* Vb  = Kb + MC;
  float* Gb  = Vb + MC;
  float* ATT = Gb + MC;
  float* FK  = ATT + MC;                     // [M, F]
  float* GN  = Kb;                           // Kb dead after wkv -> reuse

  const dim3 blk256(256);
  const dim3 gC(C_SZ / TILE_N, M_SZ / TILE_M);
  const dim3 gF(F_SZ / TILE_N, M_SZ / TILE_M);
  const dim3 gV(V_SZ / TILE_N, M_SZ / TILE_M);

  // embedding + input LN
  emb_kernel<<<(M_SZ * C_SZ) / 256, blk256, 0, stream>>>(idx, in(I_EMB), XLN, C_SZ);
  ln_kernel<<<M_SZ, blk256, 0, stream>>>(XLN, in(I_LNIN_W), in(I_LNIN_B), X, C_SZ, 1e-5f);

  for (int l = 0; l < L_SZ; ++l) {
    const int bb = 3 + l * 22;
    const float* lssA = lss + (size_t)(2 * l) * B_SZ * C_SZ;
    const float* lssF = lss + (size_t)(2 * l + 1) * B_SZ * C_SZ;

    // ---- time_mix ----
    ln_kernel<<<M_SZ, blk256, 0, stream>>>(X, in(bb + 19), in(bb + 18), XLN, C_SZ, 1e-5f);
    gemm_nt<<<gC, blk256, 0, stream>>>(XLN, in(bb + 5), Rb, in(bb + 10), lssA, T_SZ,
                                       nullptr, nullptr, M_SZ, C_SZ, C_SZ, ACT_NONE);
    gemm_nt<<<gC, blk256, 0, stream>>>(XLN, in(bb + 1), Kb, in(bb + 9), lssA, T_SZ,
                                       nullptr, nullptr, M_SZ, C_SZ, C_SZ, ACT_NONE);
    gemm_nt<<<gC, blk256, 0, stream>>>(XLN, in(bb + 12), Vb, in(bb + 11), lssA, T_SZ,
                                       nullptr, nullptr, M_SZ, C_SZ, C_SZ, ACT_NONE);
    gemm_nt<<<gC, blk256, 0, stream>>>(XLN, in(bb + 0), Gb, in(bb + 8), lssA, T_SZ,
                                       nullptr, nullptr, M_SZ, C_SZ, C_SZ, ACT_SILU);
    wkv_kernel<<<B_SZ * H_SZ, 64, 0, stream>>>(
        Rb, Kb, Vb, in(bb + 6), in(bb + 7),
        wkvS + (size_t)l * B_SZ * H_SZ * K_HD * K_HD, ATT, T_SZ, H_SZ, C_SZ);
    gn_gate<<<M_SZ * H_SZ, 64, 0, stream>>>(ATT, in(bb + 3), in(bb + 2), Gb, GN,
                                            0.00064f, H_SZ, C_SZ);
    gemm_nt<<<gC, blk256, 0, stream>>>(GN, in(bb + 4), X, nullptr, nullptr, T_SZ,
                                       nullptr, X, M_SZ, C_SZ, C_SZ, ACT_NONE);

    // ---- channel_mix ----
    ln_kernel<<<M_SZ, blk256, 0, stream>>>(X, in(bb + 21), in(bb + 20), XLN, C_SZ, 1e-5f);
    gemm_nt<<<gF, blk256, 0, stream>>>(XLN, in(bb + 13), FK, in(bb + 15), lssF, T_SZ,
                                       nullptr, nullptr, M_SZ, F_SZ, C_SZ, ACT_RELU2);
    gemm_nt<<<gC, blk256, 0, stream>>>(XLN, in(bb + 14), Rb, in(bb + 16), lssF, T_SZ,
                                       nullptr, nullptr, M_SZ, C_SZ, C_SZ, ACT_SIG);
    gemm_nt<<<gC, blk256, 0, stream>>>(FK, in(bb + 17), X, nullptr, nullptr, T_SZ,
                                       Rb, X, M_SZ, C_SZ, F_SZ, ACT_NONE);
  }

  // final LN + head projection straight into d_out
  ln_kernel<<<M_SZ, blk256, 0, stream>>>(X, in(I_LNOUT_W), in(I_LNOUT_B), XLN, C_SZ, 1e-5f);
  gemm_nt<<<gV, blk256, 0, stream>>>(XLN, in(I_HEAD), (float*)d_out, nullptr, nullptr,
                                     T_SZ, nullptr, nullptr, M_SZ, V_SZ, C_SZ, ACT_NONE);
}
